// SymQNet_56642028700125
// MI455X (gfx1250) — compile-verified, matrix operand order checked by reading, not run on hardware
//
#include <hip/hip_runtime.h>
#include <hip/hip_bf16.h>

#define NQ   64
#define LF   128
#define EE   126
#define BB   4096
#define AOUT 960

// ws layout (in _Float16 units): pre-transposed f16 weights, then X activations
#define WR_W1T   0
#define WR_W2T   32768
#define WR_N1T   49152
#define WR_N2T   81920
#define WR_ROUND 98304
#define POLWT_OFF 196608            // 960 x 256
#define XWS_OFF   442368            // 4096 x 256

typedef __attribute__((ext_vector_type(16))) _Float16 v16h;
typedef __attribute__((ext_vector_type(8)))  _Float16 v8h;
typedef __attribute__((ext_vector_type(8)))  float    v8f;

union V16U { v16h v; v8h h[2]; _Float16 e[16]; };
union V8FU { v8f v; float f[8]; };

// A/B fragment for v_wmma_f32_16x16x32_f16 from row-major f16 LDS.
// Per ISA 7.12.2: lane holds row (lane&15); lanes>=16 shifted by 8 in K.
__device__ __forceinline__ v16h frag_f16(const _Float16* rowptr, int koff, int g) {
  V16U u;
  u.h[0] = *(const v8h*)(rowptr + koff + g * 8);
  u.h[1] = *(const v8h*)(rowptr + koff + g * 8 + 16);
  return u.v;
}

__device__ __forceinline__ v8f wmma16(v16h a, v16h b, v8f c) {
  return __builtin_amdgcn_wmma_f32_16x16x32_f16(false, a, false, b, (short)0, c,
                                                false, false);
}

// Async global->LDS tile copy (CDNA5 GLOBAL_LOAD_ASYNC_TO_LDS_B128, ASYNCcnt).
// Per-lane: VDST = LDS byte offset (low 32 bits of generic shared pointer),
// VADDR = 64-bit global address. 16 B per lane per issue.
__device__ __forceinline__ void lds_copy_async(_Float16* dst, const _Float16* src,
                                               int n, int tid) {
  for (int i = tid * 8; i < n; i += 256 * 8) {
    unsigned lds_addr = (unsigned)(unsigned long long)(uintptr_t)(dst + i);
    asm volatile("global_load_async_to_lds_b128 %0, %1, off"
                 :: "v"(lds_addr), "v"(src + i) : "memory");
  }
}

__device__ __forceinline__ void async_wait() {
  asm volatile("s_wait_asynccnt 0" ::: "memory");
}

// ---------------------------------------------------------------------------
// Kernel 0: one-shot weight prep — f32 -> f16, transposed to [n][k] B layout.
// ---------------------------------------------------------------------------
__global__ __launch_bounds__(256)
void symq_wprep_kernel(const float* __restrict__ eW1, const float* __restrict__ eW2,
                       const float* __restrict__ nW1, const float* __restrict__ nW2,
                       const float* __restrict__ polW, _Float16* __restrict__ wpre)
{
  const int idx = blockIdx.x * 256 + threadIdx.x;
  const int stride = gridDim.x * 256;
  for (int k = 0; k < 2; ++k) {
    _Float16* base = wpre + k * WR_ROUND;
    const float* W1 = eW1 + k * 257 * LF;
    const float* W2 = eW2 + k * LF * LF;
    const float* N1 = nW1 + k * 256 * LF;
    const float* N2 = nW2 + k * LF * LF;
    for (int i = idx; i < 128 * 256; i += stride) {
      int n = i >> 8, kk = i & 255;
      base[WR_W1T + i] = (_Float16)W1[kk * LF + n];
    }
    for (int i = idx; i < 128 * 128; i += stride) {
      int n = i >> 7, kk = i & 127;
      base[WR_W2T + i] = (_Float16)W2[kk * LF + n];
    }
    for (int i = idx; i < 128 * 256; i += stride) {
      int n = i >> 8, kk = i & 255;
      base[WR_N1T + i] = (_Float16)N1[kk * LF + n];
    }
    for (int i = idx; i < 128 * 128; i += stride) {
      int n = i >> 7, kk = i & 127;
      base[WR_N2T + i] = (_Float16)N2[kk * LF + n];
    }
  }
  for (int i = idx; i < AOUT * 256; i += stride) {
    int n = i >> 8, kk = i & 255;
    wpre[POLWT_OFF + i] = (_Float16)polW[kk * AOUT + n];
  }
}

// ---------------------------------------------------------------------------
// Kernel 1: fully fused GNN body. One block == one batch element; all state in LDS.
// ---------------------------------------------------------------------------
__global__ __launch_bounds__(256, 1)
void symq_gnn_kernel(const float* __restrict__ z0, const int* __restrict__ src,
                     const int* __restrict__ tgt, const float* __restrict__ edge_attr,
                     const float* __restrict__ eW1, const float* __restrict__ eb1,
                     const float* __restrict__ eb2,
                     const float* __restrict__ nb1,
                     const float* __restrict__ ng,  const float* __restrict__ nbe,
                     const float* __restrict__ nb2,
                     const float* __restrict__ shW, const float* __restrict__ shb,
                     const float* __restrict__ lng, const float* __restrict__ lnb,
                     const float* __restrict__ valW, const float* __restrict__ valb,
                     const _Float16* __restrict__ wpre,
                     _Float16* __restrict__ Xws, float* __restrict__ Vout)
{
  __shared__ float                  h32[NQ * LF];     // 32 KB master node state
  __shared__ __align__(16) _Float16 h16[NQ * LF];     // 16 KB f16 mirror
  __shared__ __align__(16) _Float16 m1[128 * LF];     // 32 KB edge hidden / magg16 / u16
  __shared__ float                  magg[NQ * LF];    // 32 KB scatter target
  __shared__ float                  u32[NQ * LF];     // 32 KB node hidden
  __shared__ __align__(16) _Float16 wt[128 * 256];    // 64 KB staged weights [n][k]
  __shared__ float sb0[LF], sb1[LF], sb2[LF], sb3[LF];
  __shared__ float sgam[LF], sbet[LF];
  __shared__ int   ssrc[EE], stgt[EE];
  __shared__ float smu[NQ], srs[NQ];
  __shared__ float redA[256];
  __shared__ float redB[256];

  const int tid  = threadIdx.x;
  const int lane = tid & 31;
  const int wv   = tid >> 5;     // 8 waves
  const int b    = blockIdx.x;
  const int g    = lane >> 4;
  const int lrow = lane & 15;

  for (int i = tid; i < NQ * LF; i += 256) h32[i] = z0[b * LF + (i & (LF - 1))];
  for (int i = tid; i < EE; i += 256) { ssrc[i] = src[i]; stgt[i] = tgt[i]; }
  __syncthreads();
  const float ea = edge_attr[0];

  for (int k = 0; k < 2; ++k) {
    const float* W1  = eW1 + k * 257 * LF;   // only row 256 needed (bias fold)
    const float* B1  = eb1 + k * LF;
    const float* B2  = eb2 + k * LF;
    const float* NB1 = nb1 + k * LF;
    const float* NG  = ng  + k * LF;
    const float* NBE = nbe + k * LF;
    const float* NB2 = nb2 + k * LF;
    const _Float16* wrk = wpre + k * WR_ROUND;

    // stage: async W1t copy overlaps h16 refresh / bias loads
    lds_copy_async(wt, wrk + WR_W1T, 128 * 256, tid);
    for (int i = tid; i < NQ * LF; i += 256) { h16[i] = (_Float16)h32[i]; magg[i] = 0.f; }
    for (int i = tid; i < LF; i += 256) {
      sb0[i]  = ea * W1[256 * LF + i] + B1[i];   // edge-attr column folded into bias
      sb1[i]  = B2[i];
      sb2[i]  = NB1[i];
      sgam[i] = NG[i];  sbet[i] = NBE[i];
      sb3[i]  = NB2[i];
    }
    async_wait();
    __syncthreads();

    // ---- GEMM1 (gathered): m1 = relu([h[src] | h[tgt]] @ W1[0:256] + bias)
    {
      const int mt = wv;
      const int e  = mt * 16 + lrow;
      const int e2 = e < EE ? e : EE - 1;
      v16h afr[8];
#pragma unroll
      for (int ks = 0; ks < 8; ++ks) {
        const int node = (ks < 4) ? ssrc[e2] : stgt[e2];
        afr[ks] = frag_f16(h16 + node * LF, (ks & 3) * 32, g);
      }
      for (int nt = 0; nt < 8; ++nt) {
        v8f acc = {};
        const _Float16* wrow = wt + (nt * 16 + lrow) * 256;
#pragma unroll
        for (int ks = 0; ks < 8; ++ks)
          acc = wmma16(afr[ks], frag_f16(wrow, ks * 32, g), acc);
        const int col = nt * 16 + lrow;
        V8FU au; au.v = acc;
#pragma unroll
        for (int r = 0; r < 8; ++r) {
          const int row = mt * 16 + r + 8 * g;
          float v = au.f[r] + sb0[col];
          v = v > 0.f ? v : 0.f;
          m1[row * LF + col] = (_Float16)((row < EE) ? v : 0.f);
        }
      }
    }
    __syncthreads();

    lds_copy_async(wt, wrk + WR_W2T, 128 * 128, tid);
    async_wait();
    __syncthreads();

    // ---- GEMM2: m = m1 @ W2 + b2, scatter-add into magg via ds_add_f32
    {
      const int mt = wv;
      const _Float16* arow = m1 + (mt * 16 + lrow) * LF;
      v16h afr[4];
#pragma unroll
      for (int ks = 0; ks < 4; ++ks) afr[ks] = frag_f16(arow, ks * 32, g);
      for (int nt = 0; nt < 8; ++nt) {
        v8f acc = {};
        const _Float16* wrow = wt + (nt * 16 + lrow) * 128;
#pragma unroll
        for (int ks = 0; ks < 4; ++ks)
          acc = wmma16(afr[ks], frag_f16(wrow, ks * 32, g), acc);
        const int col = nt * 16 + lrow;
        V8FU au; au.v = acc;
#pragma unroll
        for (int r = 0; r < 8; ++r) {
          const int e = mt * 16 + r + 8 * g;
          if (e < EE) atomicAdd(&magg[ssrc[e] * LF + col], au.f[r] + sb1[col]);
        }
      }
    }
    __syncthreads();

    // bulk-convert magg -> f16 (reuse m1); async-stage NW1t concurrently
    lds_copy_async(wt, wrk + WR_N1T, 128 * 256, tid);
    for (int i = tid; i < NQ * LF; i += 256) m1[i] = (_Float16)magg[i];
    async_wait();
    __syncthreads();

    // ---- GEMM3: u = relu([h | magg] @ NW1 + nb1), M=64, N=128, K=256
    {
      const int mt   = wv & 3;
      const int node = mt * 16 + lrow;
      v16h afr[8];
#pragma unroll
      for (int ks = 0; ks < 8; ++ks) {
        if (ks < 4) afr[ks] = frag_f16(h16 + node * LF, ks * 32, g);
        else        afr[ks] = frag_f16(m1  + node * LF, (ks - 4) * 32, g);
      }
      for (int j = 0; j < 4; ++j) {
        const int nt = (wv >> 2) * 4 + j;
        v8f acc = {};
        const _Float16* wrow = wt + (nt * 16 + lrow) * 256;
#pragma unroll
        for (int ks = 0; ks < 8; ++ks)
          acc = wmma16(afr[ks], frag_f16(wrow, ks * 32, g), acc);
        const int col = nt * 16 + lrow;
        V8FU au; au.v = acc;
#pragma unroll
        for (int r = 0; r < 8; ++r) {
          const int row = mt * 16 + r + 8 * g;
          const float v = au.f[r] + sb2[col];
          u32[row * LF + col] = v > 0.f ? v : 0.f;
        }
      }
    }
    __syncthreads();

    // ---- LayerNorm stats over rows of u
    if (tid < NQ) {
      float s = 0.f;
      for (int c = 0; c < LF; ++c) s += u32[tid * LF + c];
      const float m = s * (1.f / LF);
      float v = 0.f;
      for (int c = 0; c < LF; ++c) { const float d = u32[tid * LF + c] - m; v += d * d; }
      smu[tid] = m;
      srs[tid] = rsqrtf(v * (1.f / LF) + 1e-5f);
    }
    __syncthreads();
    // LN apply -> f16 directly into m1 (u16); async-stage NW2t concurrently
    lds_copy_async(wt, wrk + WR_N2T, 128 * 128, tid);
    for (int i = tid; i < NQ * LF; i += 256) {
      const int r = i >> 7, c = i & 127;
      m1[i] = (_Float16)((u32[i] - smu[r]) * srs[r] * sgam[c] + sbet[c]);
    }
    async_wait();
    __syncthreads();

    // ---- GEMM4: h += u @ NW2 + nb2 (residual)
    {
      const int mt = wv & 3;
      const _Float16* arow = m1 + (mt * 16 + lrow) * LF;
      v16h afr[4];
#pragma unroll
      for (int ks = 0; ks < 4; ++ks) afr[ks] = frag_f16(arow, ks * 32, g);
      for (int j = 0; j < 4; ++j) {
        const int nt = (wv >> 2) * 4 + j;
        v8f acc = {};
        const _Float16* wrow = wt + (nt * 16 + lrow) * 128;
#pragma unroll
        for (int ks = 0; ks < 4; ++ks)
          acc = wmma16(afr[ks], frag_f16(wrow, ks * 32, g), acc);
        const int col = nt * 16 + lrow;
        V8FU au; au.v = acc;
#pragma unroll
        for (int r = 0; r < 8; ++r) {
          const int row = mt * 16 + r + 8 * g;
          h32[row * LF + col] += au.f[r] + sb3[col];
        }
      }
    }
    __syncthreads();
  }

  // z_G = mean over nodes
  if (tid < LF) {
    float s = 0.f;
    for (int n = 0; n < NQ; ++n) s += h32[n * LF + tid];
    redA[tid] = s * (1.f / NQ);
  }
  __syncthreads();

  // head: x = relu(z_G @ shW + shb); thread tid owns output feature tid
  float x;
  {
    float s = shb[tid];
    for (int kk = 0; kk < LF; ++kk) s += redA[kk] * shW[kk * 256 + tid];
    x = s > 0.f ? s : 0.f;
  }
  redB[tid] = x;
  __syncthreads();
  for (int s = 128; s > 0; s >>= 1) { if (tid < s) redB[tid] += redB[tid + s]; __syncthreads(); }
  const float xm = redB[0] * (1.f / 256.f);
  __syncthreads();
  const float d = x - xm;
  redB[tid] = d * d;
  __syncthreads();
  for (int s = 128; s > 0; s >>= 1) { if (tid < s) redB[tid] += redB[tid + s]; __syncthreads(); }
  const float xrs = rsqrtf(redB[0] * (1.f / 256.f) + 1e-5f);
  const float xn  = (x - xm) * xrs * lng[tid] + lnb[tid];
  Xws[b * 256 + tid] = (_Float16)xn;
  __syncthreads();
  redB[tid] = xn * valW[tid];
  __syncthreads();
  for (int s = 128; s > 0; s >>= 1) { if (tid < s) redB[tid] += redB[tid + s]; __syncthreads(); }
  if (tid == 0) Vout[b] = redB[0] + valb[0];
}

// ---------------------------------------------------------------------------
// Kernel 2: logits = X(4096x256) @ pol_W(256x960) + pol_b  — WMMA tiled GEMM.
// Operands staged from pre-packed f16 via async global->LDS copies.
// ---------------------------------------------------------------------------
__global__ __launch_bounds__(256, 2)
void symq_head_kernel(const _Float16* __restrict__ Xws, const _Float16* __restrict__ polWt,
                      const float* __restrict__ polb, float* __restrict__ logits)
{
  __shared__ __align__(16) _Float16 xt[64 * 256];   // 32 KB
  __shared__ __align__(16) _Float16 wtt[64 * 256];  // 32 KB [n][k]
  const int tid  = threadIdx.x;
  const int lane = tid & 31;
  const int wv   = tid >> 5;
  const int g    = lane >> 4;
  const int lrow = lane & 15;
  const int m0   = blockIdx.x * 64;
  const int n0   = blockIdx.y * 64;

  lds_copy_async(xt,  Xws   + m0 * 256, 64 * 256, tid);
  lds_copy_async(wtt, polWt + n0 * 256, 64 * 256, tid);
  async_wait();
  __syncthreads();

  const int mt = wv & 3;
  const _Float16* arow = xt + (mt * 16 + lrow) * 256;
  v16h afr[8];
#pragma unroll
  for (int ks = 0; ks < 8; ++ks) afr[ks] = frag_f16(arow, ks * 32, g);

  for (int j = 0; j < 2; ++j) {
    const int nt = (wv >> 2) * 2 + j;
    v8f acc = {};
    const _Float16* wrow = wtt + (nt * 16 + lrow) * 256;
#pragma unroll
    for (int ks = 0; ks < 8; ++ks)
      acc = wmma16(afr[ks], frag_f16(wrow, ks * 32, g), acc);
    const int col  = n0 + nt * 16 + lrow;
    const float bb = polb[col];
    V8FU au; au.v = acc;
#pragma unroll
    for (int r = 0; r < 8; ++r) {
      const int row = m0 + mt * 16 + r + 8 * g;
      logits[row * AOUT + col] = au.f[r] + bb;
    }
  }
}

extern "C" void kernel_launch(void* const* d_in, const int* in_sizes, int n_in,
                              void* d_out, int out_size, void* d_ws, size_t ws_size,
                              hipStream_t stream) {
  const float* z0   = (const float*)d_in[0];
  const int*   src  = (const int*)d_in[1];
  const int*   tgt  = (const int*)d_in[2];
  const float* ea   = (const float*)d_in[3];
  const float* eW1  = (const float*)d_in[4];
  const float* eb1  = (const float*)d_in[5];
  const float* eW2  = (const float*)d_in[6];
  const float* eb2  = (const float*)d_in[7];
  const float* nW1  = (const float*)d_in[8];
  const float* nb1  = (const float*)d_in[9];
  const float* ngm  = (const float*)d_in[10];
  const float* nbe  = (const float*)d_in[11];
  const float* nW2  = (const float*)d_in[12];
  const float* nb2  = (const float*)d_in[13];
  const float* shW  = (const float*)d_in[14];
  const float* shb  = (const float*)d_in[15];
  const float* lng  = (const float*)d_in[16];
  const float* lnb  = (const float*)d_in[17];
  const float* polW = (const float*)d_in[18];
  const float* polb = (const float*)d_in[19];
  const float* valW = (const float*)d_in[20];
  const float* valb = (const float*)d_in[21];

  float*    logits = (float*)d_out;                       // (4096, 960)
  float*    Vout   = (float*)d_out + (size_t)BB * AOUT;   // (4096,)
  _Float16* wpre   = (_Float16*)d_ws;                     // packed f16 weights
  _Float16* Xws    = wpre + XWS_OFF;                      // (4096, 256) activations

  symq_wprep_kernel<<<dim3(256), dim3(256), 0, stream>>>(
      eW1, eW2, nW1, nW2, polW, wpre);

  symq_gnn_kernel<<<dim3(BB), dim3(256), 0, stream>>>(
      z0, src, tgt, ea, eW1, eb1, eb2, nb1, ngm, nbe, nb2,
      shW, shb, lng, lnb, valW, valb, wpre, Xws, Vout);

  symq_head_kernel<<<dim3(BB / 64, AOUT / 64), dim3(256), 0, stream>>>(
      Xws, wpre + POLWT_OFF, polb, logits);
}
